// GCNDecoder_867583394201
// MI455X (gfx1250) — matrix operand order repaired and checked by reading, not compiled, not run on hardware
//
#include <hip/hip_runtime.h>
#include <hip/hip_bf16.h>

// ---------- types ----------
typedef __bf16 bf16_t;
typedef bf16_t v16bf __attribute__((ext_vector_type(16)));
typedef float  v8f   __attribute__((ext_vector_type(8)));
typedef unsigned int u32x4 __attribute__((ext_vector_type(4)));

union BF16x16 { v16bf v; u32x4 q[2]; };

static __device__ __forceinline__ unsigned short f2bf(float f) {
    union { float f; unsigned u; } v; v.f = f;
    unsigned u = v.u;
    unsigned r = u + 0x7FFFu + ((u >> 16) & 1u);   // round-to-nearest-even
    return (unsigned short)(r >> 16);
}
static __device__ __forceinline__ float bf2f(unsigned short h) {
    union { unsigned u; float f; } v; v.u = ((unsigned)h) << 16;
    return v.f;
}

#define N_NODES 40000
#define N_EDGES 640000
#define BATCH   2
#define HID     128
#define ROWS    (BATCH * N_NODES)   // 80000

// ---------- 1) initial linear: x[r][c] = sum_i p[r][i]*W[i][c] + b[c] ----------
__global__ void k_init_linear(const float* __restrict__ p,
                              const float* __restrict__ W,
                              const float* __restrict__ b,
                              float* __restrict__ x) {
    int tid = blockIdx.x * blockDim.x + threadIdx.x;
    int r  = tid >> 5;
    int c0 = (tid & 31) * 4;
    if (r >= ROWS) return;
    float p0 = p[r * 3 + 0], p1 = p[r * 3 + 1], p2 = p[r * 3 + 2];
    float4 w0 = *(const float4*)(W + 0 * HID + c0);
    float4 w1 = *(const float4*)(W + 1 * HID + c0);
    float4 w2 = *(const float4*)(W + 2 * HID + c0);
    float4 bb = *(const float4*)(b + c0);
    float4 o;
    o.x = p0 * w0.x + p1 * w1.x + p2 * w2.x + bb.x;
    o.y = p0 * w0.y + p1 * w1.y + p2 * w2.y + bb.y;
    o.z = p0 * w0.z + p1 * w1.z + p2 * w2.z + bb.z;
    o.w = p0 * w0.w + p1 * w1.w + p2 * w2.w + bb.w;
    *(float4*)(x + (size_t)r * HID + c0) = o;
}

// ---------- 2) relu + f32 -> bf16 ----------
__global__ void k_relu_cvt(const float* __restrict__ in,
                           unsigned short* __restrict__ out, int nquads) {
    int tid = blockIdx.x * blockDim.x + threadIdx.x;
    if (tid >= nquads) return;
    size_t i0 = (size_t)tid * 4;
    float4 v = *(const float4*)(in + i0);
    v.x = fmaxf(v.x, 0.f); v.y = fmaxf(v.y, 0.f);
    v.z = fmaxf(v.z, 0.f); v.w = fmaxf(v.w, 0.f);
    uint2 o;
    o.x = (unsigned)f2bf(v.x) | ((unsigned)f2bf(v.y) << 16);
    o.y = (unsigned)f2bf(v.z) | ((unsigned)f2bf(v.w) << 16);
    *(uint2*)(out + i0) = o;
}

// ---------- 3) f32 -> bf16 (no relu), for spmm result ----------
__global__ void k_cvt(const float* __restrict__ in,
                      unsigned short* __restrict__ out, int nquads) {
    int tid = blockIdx.x * blockDim.x + threadIdx.x;
    if (tid >= nquads) return;
    size_t i0 = (size_t)tid * 4;
    float4 v = *(const float4*)(in + i0);
    uint2 o;
    o.x = (unsigned)f2bf(v.x) | ((unsigned)f2bf(v.y) << 16);
    o.y = (unsigned)f2bf(v.z) | ((unsigned)f2bf(v.w) << 16);
    *(uint2*)(out + i0) = o;
}

// ---------- 4) edge scatter: s[row] += val * h[col]  (one wave per edge/batch) ----------
__global__ void k_scatter(const unsigned short* __restrict__ h,
                          const int* __restrict__ er,
                          const int* __restrict__ ec,
                          const float* __restrict__ av,
                          float* __restrict__ s) {
    int tid  = blockIdx.x * blockDim.x + threadIdx.x;
    int slot = tid >> 5;
    if (slot >= BATCH * N_EDGES) return;
    int lane = tid & 31;
    int b = slot / N_EDGES;
    int e = slot - b * N_EDGES;
    int row = er[e], col = ec[e];
    float val = av[e];
    const unsigned short* hp = h + ((size_t)(b * N_NODES + col)) * HID + lane * 4;
    uint2 d = *(const uint2*)hp;
    float f0 = bf2f((unsigned short)(d.x & 0xffff));
    float f1 = bf2f((unsigned short)(d.x >> 16));
    float f2 = bf2f((unsigned short)(d.y & 0xffff));
    float f3 = bf2f((unsigned short)(d.y >> 16));
    float* sp = s + ((size_t)(b * N_NODES + row)) * HID + lane * 4;
    atomicAdd(sp + 0, val * f0);
    atomicAdd(sp + 1, val * f1);
    atomicAdd(sp + 2, val * f2);
    atomicAdd(sp + 3, val * f3);
}

// ---------- 5) fused WMMA GEMM: out = h@Ws + s@Wf + (bs+bf) [+ res] ----------
// One wave computes a 16-row x (NTILES*16)-col tile.  K = 128 (4 steps of 32).
// Weights converted fp32 -> bf16, stored TRANSPOSED [n][k] in LDS (KPAD pad).
template <int NTILES>
__launch_bounds__(256) __global__
void k_gemm(const unsigned short* __restrict__ h,
            const unsigned short* __restrict__ s,
            const float* __restrict__ Ws, const float* __restrict__ Wf,
            const float* __restrict__ bs, const float* __restrict__ bf,
            const float* __restrict__ res, float* __restrict__ out,
            int NC) {
    constexpr int KPAD = 136;               // 128 + 8 halves: 16B-aligned rows, no bank clash
    constexpr int NLDS = NTILES * 16 * KPAD;
    __shared__ __attribute__((aligned(16))) unsigned short lds[2 * NLDS];

    int tid = threadIdx.x;
    // zero LDS (covers zero-padding for NC < NTILES*16)
    for (int i = tid; i < NLDS; i += 256) ((unsigned*)lds)[i] = 0u;
    __syncthreads();
    // fill transposed bf16 weights: lds[m][n*KPAD + k] = W_m[k][n]
    int total = HID * NC;
    for (int i = tid; i < total; i += 256) {
        int k = i / NC;
        int n = i - k * NC;
        lds[n * KPAD + k]        = f2bf(Ws[i]);
        lds[NLDS + n * KPAD + k] = f2bf(Wf[i]);
    }
    __syncthreads();

    int wave = tid >> 5, lane = tid & 31;
    int tile = blockIdx.x * 8 + wave;
    int rowbase = tile * 16;
    if (rowbase >= ROWS) return;            // wave-uniform; EXEC stays all-ones
    int g  = lane >> 4;                     // half-wave group
    int cl = lane & 15;

    const unsigned short* hrow = h + (size_t)(rowbase + cl) * HID;
    const unsigned short* srow = s + (size_t)(rowbase + cl) * HID;

    v8f acc[NTILES];
#pragma unroll
    for (int t = 0; t < NTILES; ++t) {
        int col = t * 16 + cl;
        float bias = (col < NC) ? (bs[col] + bf[col]) : 0.f;
#pragma unroll
        for (int r = 0; r < 8; ++r) acc[t][r] = bias;
    }

#pragma unroll
    for (int ks = 0; ks < 4; ++ks) {
        // A fragment, this code's contiguous k order (same as B): lane<16 -> K = ks*32 + {0..15},
        // lane>=16 -> K = ks*32 + {16..31}.  Two b128 loads each.
        int c0 = ks * 32 + g * 16;
        BF16x16 ah, as;
        ah.q[0] = *(const u32x4*)(hrow + c0);
        ah.q[1] = *(const u32x4*)(hrow + c0 + 8);
        as.q[0] = *(const u32x4*)(srow + c0);
        as.q[1] = *(const u32x4*)(srow + c0 + 8);

        int kb = ks * 32 + g * 16;          // B fragment: lanes 0-15 K=kb..kb+15
#pragma unroll
        for (int t = 0; t < NTILES; ++t) {
            int n = t * 16 + cl;
            BF16x16 bS, bF;
            const u32x4* pS = (const u32x4*)&lds[n * KPAD + kb];
            bS.q[0] = pS[0]; bS.q[1] = pS[1];
            acc[t] = __builtin_amdgcn_wmma_f32_16x16x32_bf16(
                false, ah.v, false, bS.v, (short)0, acc[t], false, false);
            const u32x4* pF = (const u32x4*)&lds[NLDS + n * KPAD + kb];
            bF.q[0] = pF[0]; bF.q[1] = pF[1];
            acc[t] = __builtin_amdgcn_wmma_f32_16x16x32_bf16(
                false, as.v, false, bF.v, (short)0, acc[t], false, false);
        }
    }

    // store C layout: lane cl = col, VGPR r -> row = rowbase + r + 8*g
#pragma unroll
    for (int t = 0; t < NTILES; ++t) {
        int col = t * 16 + cl;
        if (col < NC) {
#pragma unroll
            for (int r = 0; r < 8; ++r) {
                int row = rowbase + r + 8 * g;
                size_t idx = (size_t)row * NC + col;
                float v = acc[t][r];
                if (res) v += res[idx];
                out[idx] = v;
            }
        }
    }
}

// ---------- host orchestration ----------
extern "C" void kernel_launch(void* const* d_in, const int* in_sizes, int n_in,
                              void* d_out, int out_size, void* d_ws, size_t ws_size,
                              hipStream_t stream) {
    const float* p    = (const float*)d_in[0];
    const float* av   = (const float*)d_in[1];
    const int*   er   = (const int*)d_in[2];
    const int*   ec   = (const int*)d_in[3];
    const float* Wi   = (const float*)d_in[4];
    const float* bi   = (const float*)d_in[5];
    const float* Wfb  = (const float*)d_in[6];
    const float* bfb  = (const float*)d_in[7];
    const float* Wsb  = (const float*)d_in[8];
    const float* bsb  = (const float*)d_in[9];
    const float* Wfo  = (const float*)d_in[10];
    const float* bfo  = (const float*)d_in[11];
    const float* Wso  = (const float*)d_in[12];
    const float* bso  = (const float*)d_in[13];

    const size_t NF   = (size_t)ROWS * HID;          // 10,240,000 f32 elements
    const size_t SZ_F = NF * sizeof(float);          // 40,960,000 B
    const size_t SZ_H = NF * sizeof(unsigned short); // 20,480,000 B

    char* ws = (char*)d_ws;
    float*          x   = (float*)(ws);
    float*          net = (float*)(ws + SZ_F);
    float*          s32 = (float*)(ws + 2 * SZ_F);
    unsigned short* h   = (unsigned short*)(ws + 3 * SZ_F);
    unsigned short* sb  = (unsigned short*)(ws + 3 * SZ_F + SZ_H);

    const int nquads = (int)(NF / 4);                // 2,560,000
    const int CB = 256;

    // x = p @ W_init + b_init
    k_init_linear<<<(ROWS * 32 + CB - 1) / CB, CB, 0, stream>>>(p, Wi, bi, x);

    auto gconv = [&](const float* xin, const float* Wf, const float* bfp,
                     const float* Ws, const float* bsp,
                     int NC, float* outp, const float* resp) {
        k_relu_cvt<<<(nquads + CB - 1) / CB, CB, 0, stream>>>(xin, h, nquads);
        hipMemsetAsync(s32, 0, SZ_F, stream);
        k_scatter<<<(BATCH * N_EDGES * 32) / CB, CB, 0, stream>>>(h, er, ec, av, s32);
        k_cvt<<<(nquads + CB - 1) / CB, CB, 0, stream>>>(s32, sb, nquads);
        int blocks = (ROWS / 16 + 7) / 8;            // 625
        if (NC == HID)
            k_gemm<8><<<blocks, 256, 0, stream>>>(h, sb, Ws, Wf, bsp, bfp, resp, outp, NC);
        else
            k_gemm<1><<<blocks, 256, 0, stream>>>(h, sb, Ws, Wf, bsp, bfp, resp, outp, NC);
    };

    const size_t WSTEP = (size_t)HID * HID;          // 16384
    // Block 1
    gconv(x,   Wfb + 0 * WSTEP, bfb + 0 * HID, Wsb + 0 * WSTEP, bsb + 0 * HID, HID, net, nullptr);
    gconv(net, Wfb + 1 * WSTEP, bfb + 1 * HID, Wsb + 1 * WSTEP, bsb + 1 * HID, HID, x,   x);
    // Block 2
    gconv(x,   Wfb + 2 * WSTEP, bfb + 2 * HID, Wsb + 2 * WSTEP, bsb + 2 * HID, HID, net, nullptr);
    gconv(net, Wfb + 3 * WSTEP, bfb + 3 * HID, Wsb + 3 * WSTEP, bsb + 3 * HID, HID, x,   x);
    // Final 128 -> 3
    gconv(x, Wfo, bfo, Wso, bso, 3, (float*)d_out, nullptr);
}
